// HAKGModel_31628139168308
// MI455X (gfx1250) — matrix operand order repaired and checked by reading, not compiled, not run on hardware
//
#include <hip/hip_runtime.h>
#include <hip/hip_bf16.h>

#define N_USERS    100000
#define N_ITEMS    50000
#define N_ENTITIES 200000
#define EMB        64
#define MARGIN_CCL 0.8f
#define NUM_NEG    64
#define ANGLE_W    0.1f
#define ANGLE_DROP 0.5f
#define N_EDGES    1000000
#define NNZ        1000000
#define N_TRIPLETS 200000
#define BATCH      4096
#define EPSF       1e-6f

typedef __attribute__((ext_vector_type(16))) _Float16 v16h;
typedef __attribute__((ext_vector_type(8)))  float    v8f;

__device__ __forceinline__ float wave_sum(float v) {
  #pragma unroll
  for (int off = 16; off; off >>= 1) v += __shfl_xor(v, off, 32);
  return v;
}

// ---------------- GCN scatter kernels (one wave per edge, float2 per lane) ----

// entity_agg[head] += entity_cur[tail] * rel_weight[etype-1]
__global__ void edge_scatter_kernel(const float* __restrict__ ent_cur,
                                    const float* __restrict__ relw,
                                    const int* __restrict__ head,
                                    const int* __restrict__ tail,
                                    const int* __restrict__ etype,
                                    float* __restrict__ ent_agg, int nE) {
  long long tid = (long long)blockIdx.x * blockDim.x + threadIdx.x;
  int e = (int)(tid >> 5);
  int lane = threadIdx.x & 31;
  if (e >= nE) return;
  int h = head[e], t = tail[e], r = etype[e] - 1;
  float2 ev = ((const float2*)(ent_cur + (long long)t * EMB))[lane];
  float2 rv = ((const float2*)(relw    + (long long)r * EMB))[lane];
  float* dst = ent_agg + (long long)h * EMB + lane * 2;
  atomicAdd(dst,     ev.x * rv.x);
  atomicAdd(dst + 1, ev.y * rv.y);
}

// usr_agg[rows] += vals*ent_cur[cols]; ucf_agg[rows] += vals*cf_cur[cols]
__global__ void interact_scatter_kernel(const float* __restrict__ ent_cur,
                                        const float* __restrict__ cf_cur,
                                        const float* __restrict__ vals,
                                        const int* __restrict__ rows,
                                        const int* __restrict__ cols,
                                        float* __restrict__ usr_agg,
                                        float* __restrict__ ucf_agg, int nnz) {
  long long tid = (long long)blockIdx.x * blockDim.x + threadIdx.x;
  int e = (int)(tid >> 5);
  int lane = threadIdx.x & 31;
  if (e >= nnz) return;
  float v = vals[e];
  int r = rows[e], c = cols[e];
  float2 ev = ((const float2*)(ent_cur + (long long)c * EMB))[lane];
  float2 cv = ((const float2*)(cf_cur  + (long long)c * EMB))[lane];
  float* du = usr_agg + (long long)r * EMB + lane * 2;
  atomicAdd(du,     v * ev.x);
  atomicAdd(du + 1, v * ev.y);
  float* dc = ucf_agg + (long long)r * EMB + lane * 2;
  atomicAdd(dc,     v * cv.x);
  atomicAdd(dc + 1, v * cv.y);
}

// cf_agg[cols] += vals * ucf[rows]
__global__ void item_scatter_kernel(const float* __restrict__ ucf,
                                    const float* __restrict__ vals,
                                    const int* __restrict__ rows,
                                    const int* __restrict__ cols,
                                    float* __restrict__ cf_agg, int nnz) {
  long long tid = (long long)blockIdx.x * blockDim.x + threadIdx.x;
  int e = (int)(tid >> 5);
  int lane = threadIdx.x & 31;
  if (e >= nnz) return;
  float v = vals[e];
  int r = rows[e], c = cols[e];
  float2 uv = ((const float2*)(ucf + (long long)r * EMB))[lane];
  float* d = cf_agg + (long long)c * EMB + lane * 2;
  atomicAdd(d,     v * uv.x);
  atomicAdd(d + 1, v * uv.y);
}

// L2-normalize rows of agg in place; res += normalized. One wave per row.
__global__ void norm_acc_kernel(float* __restrict__ agg, float* __restrict__ res, int nrows) {
  long long tid = (long long)blockIdx.x * blockDim.x + threadIdx.x;
  int row = (int)(tid >> 5);
  int lane = threadIdx.x & 31;
  if (row >= nrows) return;
  float2* rp = (float2*)(agg + (long long)row * EMB);
  float2 v = rp[lane];
  float ss = wave_sum(v.x * v.x + v.y * v.y);
  float inv = 1.0f / fmaxf(sqrtf(ss), 1e-12f);
  v.x *= inv; v.y *= inv;
  rp[lane] = v;
  float2* op = (float2*)(res + (long long)row * EMB);
  float2 o = op[lane];
  o.x += v.x; o.y += v.y;
  op[lane] = o;
}

// dst = l2norm(rows of src)
__global__ void norm_out_kernel(const float* __restrict__ src, float* __restrict__ dst, int nrows) {
  long long tid = (long long)blockIdx.x * blockDim.x + threadIdx.x;
  int row = (int)(tid >> 5);
  int lane = threadIdx.x & 31;
  if (row >= nrows) return;
  float2 v = ((const float2*)(src + (long long)row * EMB))[lane];
  float ss = wave_sum(v.x * v.x + v.y * v.y);
  float inv = 1.0f / fmaxf(sqrtf(ss), 1e-12f);
  v.x *= inv; v.y *= inv;
  ((float2*)(dst + (long long)row * EMB))[lane] = v;
}

// ---------------- Contrastive loss: one wave per batch element, WMMA dots ----
//
// Per batch element: 64 KG-negative dots + 64 CF-negative dots, K=64 each.
// Group of 16 negatives -> A matrix (16x32 f16 per chunk), user vector broadcast
// into all 16 columns of B. Two v_wmma_f32_16x16x32_f16 accumulate K=64.
// All D columns are identical; lanes 0-15 carry rows 0-7, lanes 16-31 rows 8-15.
__global__ void contrastive_kernel(const float* __restrict__ usr_n,
                                   const float* __restrict__ ent_n,
                                   const float* __restrict__ cf_n,
                                   const int* __restrict__ user,
                                   const int* __restrict__ pos,
                                   const int* __restrict__ neg,
                                   float* __restrict__ out) {
  long long tid = (long long)blockIdx.x * blockDim.x + threadIdx.x;
  int b = (int)(tid >> 5);
  int lane = threadIdx.x & 31;
  if (b >= BATCH) return;  // wave-uniform; grid covers exactly BATCH waves

  const float* u   = usr_n + (long long)user[b] * EMB;
  const float* p   = ent_n + (long long)pos[b]  * EMB;
  const float* pcf = cf_n  + (long long)pos[b]  * EMB;

  // ui_pos = relu(2 - (u.p + u.pcf))
  float2 uu = ((const float2*)u)[lane];
  float2 pp = ((const float2*)p)[lane];
  float2 pc = ((const float2*)pcf)[lane];
  float part = wave_sum(uu.x * pp.x + uu.y * pp.y + uu.x * pc.x + uu.y * pc.y);
  float ui_pos = fmaxf(0.0f, 2.0f - part);

  // Build B operand (user vector in every column). B k order used here (contiguous, matched by A):
  // lanes 0-15 hold K=0..15 of the chunk, lanes 16-31 hold K=16..31.
  int boff = (lane < 16) ? 0 : 16;
  v16h b0, b1;
  #pragma unroll
  for (int i = 0; i < 16; ++i) {
    b0[i] = (_Float16)u[boff + i];
    b1[i] = (_Float16)u[32 + boff + i];
  }

  // A k order used here (same as B): lane l holds row l%16; lanes<16: K {0..15}, lanes>=16: K {16..31}
  int koff = (lane < 16) ? 0 : 16;

  float negloss = 0.0f;
  #pragma unroll
  for (int s = 0; s < 2; ++s) {
    const float* tab = (s == 0) ? ent_n : cf_n;
    float ssum = 0.0f, scnt = 0.0f;
    #pragma unroll
    for (int g = 0; g < 4; ++g) {
      int ni = neg[(long long)b * NUM_NEG + g * 16 + (lane & 15)];
      const float* nr = tab + (long long)ni * EMB;
      v16h a0, a1;
      #pragma unroll
      for (int i = 0; i < 8; ++i) {
        a0[i]     = (_Float16)nr[koff + i];
        a0[8 + i] = (_Float16)nr[koff + 8 + i];
        a1[i]     = (_Float16)nr[32 + koff + i];
        a1[8 + i] = (_Float16)nr[32 + koff + 8 + i];
      }
      v8f acc = {};
      acc = __builtin_amdgcn_wmma_f32_16x16x32_f16(false, a0, false, b0,
                                                   (short)0, acc, false, false);
      acc = __builtin_amdgcn_wmma_f32_16x16x32_f16(false, a1, false, b1,
                                                   (short)0, acc, false, false);
      #pragma unroll
      for (int i = 0; i < 8; ++i) {
        float sv = acc[i] - MARGIN_CCL;
        if (sv > 0.0f) { ssum += sv; scnt += 1.0f; }
      }
    }
    // lanes<16 hold rows 0-7 (all columns identical), lanes>=16 hold rows 8-15
    float tsum = ssum + __shfl_xor(ssum, 16, 32);
    float tcnt = scnt + __shfl_xor(scnt, 16, 32);
    negloss += tsum / (tcnt + 1e-5f);
  }

  if (lane == 0) atomicAdd(out, (ui_pos + negloss) * (1.0f / BATCH));
}

// ---------------- Angle loss: one wave per triplet ---------------------------
__global__ void angle_kernel(const float* __restrict__ ent0,
                             const int* __restrict__ th,
                             const int* __restrict__ tt,
                             float* __restrict__ out, int n) {
  long long tid = (long long)blockIdx.x * blockDim.x + threadIdx.x;
  int t = (int)(tid >> 5);
  int lane = threadIdx.x & 31;
  if (t >= n) return;
  float2 hv = ((const float2*)(ent0 + (long long)th[t] * EMB))[lane];
  float2 tv = ((const float2*)(ent0 + (long long)tt[t] * EMB))[lane];
  hv.x *= ANGLE_DROP; hv.y *= ANGLE_DROP;
  tv.x *= ANGLE_DROP; tv.y *= ANGLE_DROP;
  float nu2 = hv.x * hv.x + hv.y * hv.y;
  float nv2 = tv.x * tv.x + tv.y * tv.y;
  float dp  = hv.x * tv.x + hv.y * tv.y;
  float dx = hv.x - tv.x, dy = hv.y - tv.y;
  float ed2 = dx * dx + dy * dy;
  #pragma unroll
  for (int off = 16; off; off >>= 1) {
    nu2 += __shfl_xor(nu2, off, 32);
    nv2 += __shfl_xor(nv2, off, 32);
    dp  += __shfl_xor(dp,  off, 32);
    ed2 += __shfl_xor(ed2, off, 32);
  }
  if (lane == 0) {
    float nu = sqrtf(nu2), ed = sqrtf(ed2);
    float num = dp * (1.0f + nu2) - nu2 * (1.0f + nv2);
    float den = nu * ed * sqrtf(fmaxf(1.0f + nv2 * nu2 - 2.0f * dp, EPSF)) + EPSF;
    float ca = fminf(fmaxf(num / den, -1.0f + EPSF), 1.0f - EPSF);
    float angle = acosf(ca);
    float sqnu = fminf(fmaxf(nu2, 0.0f), 1.0f - EPSF);
    float sa = fminf(fmaxf(0.1f * (1.0f - sqnu) / sqrtf(sqnu), -1.0f + EPSF), 1.0f - EPSF);
    float half_ap = asinf(sa);
    float v = fmaxf(angle - half_ap, 0.0f);
    atomicAdd(out, v * (ANGLE_W / (float)N_TRIPLETS));
  }
}

// ---------------- Host side ---------------------------------------------------

static inline int blocksFor(long long threads) { return (int)((threads + 255) / 256); }

extern "C" void kernel_launch(void* const* d_in, const int* in_sizes, int n_in,
                              void* d_out, int out_size, void* d_ws, size_t ws_size,
                              hipStream_t stream) {
  const float* all_embed = (const float*)d_in[0];
  const float* item_cf0  = (const float*)d_in[1];
  const float* relw      = (const float*)d_in[2];
  const float* vals      = (const float*)d_in[3];
  const int*   user      = (const int*)d_in[4];
  const int*   pos       = (const int*)d_in[5];
  const int*   neg       = (const int*)d_in[6];
  const int*   head      = (const int*)d_in[7];
  const int*   tail      = head + N_EDGES;
  const int*   etype     = (const int*)d_in[8];
  const int*   irows     = (const int*)d_in[9];
  const int*   icols     = (const int*)d_in[10];
  const int*   th        = (const int*)d_in[11];
  const int*   tt        = (const int*)d_in[12];

  const float* usr0 = all_embed;
  const float* ent0 = all_embed + (size_t)N_USERS * EMB;

  char* w = (char*)d_ws;
  auto alloc = [&](size_t nf) -> float* { float* p = (float*)w; w += nf * sizeof(float); return p; };
  float* ent_s0  = alloc((size_t)N_ENTITIES * EMB);
  float* ent_s1  = alloc((size_t)N_ENTITIES * EMB);
  float* ent_res = alloc((size_t)N_ENTITIES * EMB);
  float* usr_s0  = alloc((size_t)N_USERS * EMB);
  float* usr_s1  = alloc((size_t)N_USERS * EMB);
  float* usr_res = alloc((size_t)N_USERS * EMB);
  float* cf_s0   = alloc((size_t)N_ITEMS * EMB);
  float* cf_s1   = alloc((size_t)N_ITEMS * EMB);
  float* cf_res  = alloc((size_t)N_ITEMS * EMB);
  float* ucf_tmp = alloc((size_t)N_USERS * EMB);

  const size_t entB = (size_t)N_ENTITIES * EMB * sizeof(float);
  const size_t usrB = (size_t)N_USERS * EMB * sizeof(float);
  const size_t cfB  = (size_t)N_ITEMS * EMB * sizeof(float);

  // residual accumulators start at the raw embeddings
  hipMemcpyAsync(ent_res, ent0,     entB, hipMemcpyDeviceToDevice, stream);
  hipMemcpyAsync(usr_res, usr0,     usrB, hipMemcpyDeviceToDevice, stream);
  hipMemcpyAsync(cf_res,  item_cf0, cfB,  hipMemcpyDeviceToDevice, stream);

  // ---------------- hop 1 (read original embeddings, scatter into s0) --------
  hipMemsetAsync(ent_s0, 0, entB, stream);
  hipMemsetAsync(usr_s0, 0, usrB, stream);
  hipMemsetAsync(cf_s0,  0, cfB,  stream);
  hipMemsetAsync(ucf_tmp, 0, usrB, stream);

  edge_scatter_kernel<<<blocksFor((long long)N_EDGES * 32), 256, 0, stream>>>(
      ent0, relw, head, tail, etype, ent_s0, N_EDGES);
  interact_scatter_kernel<<<blocksFor((long long)NNZ * 32), 256, 0, stream>>>(
      ent0, item_cf0, vals, irows, icols, usr_s0, ucf_tmp, NNZ);
  item_scatter_kernel<<<blocksFor((long long)NNZ * 32), 256, 0, stream>>>(
      ucf_tmp, vals, irows, icols, cf_s0, NNZ);

  norm_acc_kernel<<<blocksFor((long long)N_ENTITIES * 32), 256, 0, stream>>>(ent_s0, ent_res, N_ENTITIES);
  norm_acc_kernel<<<blocksFor((long long)N_USERS * 32), 256, 0, stream>>>(usr_s0, usr_res, N_USERS);
  norm_acc_kernel<<<blocksFor((long long)N_ITEMS * 32), 256, 0, stream>>>(cf_s0, cf_res, N_ITEMS);

  // ---------------- hop 2 (read s0, scatter into s1) -------------------------
  hipMemsetAsync(ent_s1, 0, entB, stream);
  hipMemsetAsync(usr_s1, 0, usrB, stream);
  hipMemsetAsync(cf_s1,  0, cfB,  stream);
  hipMemsetAsync(ucf_tmp, 0, usrB, stream);

  edge_scatter_kernel<<<blocksFor((long long)N_EDGES * 32), 256, 0, stream>>>(
      ent_s0, relw, head, tail, etype, ent_s1, N_EDGES);
  interact_scatter_kernel<<<blocksFor((long long)NNZ * 32), 256, 0, stream>>>(
      ent_s0, cf_s0, vals, irows, icols, usr_s1, ucf_tmp, NNZ);
  item_scatter_kernel<<<blocksFor((long long)NNZ * 32), 256, 0, stream>>>(
      ucf_tmp, vals, irows, icols, cf_s1, NNZ);

  norm_acc_kernel<<<blocksFor((long long)N_ENTITIES * 32), 256, 0, stream>>>(ent_s1, ent_res, N_ENTITIES);
  norm_acc_kernel<<<blocksFor((long long)N_USERS * 32), 256, 0, stream>>>(usr_s1, usr_res, N_USERS);
  norm_acc_kernel<<<blocksFor((long long)N_ITEMS * 32), 256, 0, stream>>>(cf_s1, cf_res, N_ITEMS);

  // ---------------- normalized residuals for the contrastive loss ------------
  norm_out_kernel<<<blocksFor((long long)N_ENTITIES * 32), 256, 0, stream>>>(ent_res, ent_s0, N_ENTITIES);
  norm_out_kernel<<<blocksFor((long long)N_USERS * 32), 256, 0, stream>>>(usr_res, usr_s0, N_USERS);
  norm_out_kernel<<<blocksFor((long long)N_ITEMS * 32), 256, 0, stream>>>(cf_res, cf_s0, N_ITEMS);

  // ---------------- losses ---------------------------------------------------
  hipMemsetAsync(d_out, 0, (size_t)out_size * sizeof(float), stream);

  contrastive_kernel<<<blocksFor((long long)BATCH * 32), 256, 0, stream>>>(
      usr_s0, ent_s0, cf_s0, user, pos, neg, (float*)d_out);

  angle_kernel<<<blocksFor((long long)N_TRIPLETS * 32), 256, 0, stream>>>(
      ent0, th, tt, (float*)d_out, N_TRIPLETS);
}
